// Model_24558622999077
// MI455X (gfx1250) — compile-verified, matrix operand order checked by reading, not run on hardware
//
#include <hip/hip_runtime.h>
#include <hip/hip_bf16.h>
#include <math.h>
#include <stdint.h>

// ---- problem constants (from reference) ----
#define BB   16
#define LL   384
#define DM   512
#define DFF  2048
#define CIN  21
#define TKK  5        // int(log(384)) = 5
#define LABL 192
#define PRD  192

typedef __attribute__((ext_vector_type(16))) __bf16 v16bf;
typedef __attribute__((ext_vector_type(8)))  float  v8f;

#if defined(__AMDGCN__) && __has_builtin(__builtin_amdgcn_tensor_load_to_lds) && __has_builtin(__builtin_amdgcn_s_wait_tensorcnt)
#define HAVE_TDM 1
#endif

#ifdef HAVE_TDM
typedef __attribute__((ext_vector_type(4))) unsigned int v4u;
typedef __attribute__((ext_vector_type(8))) int v8i;
typedef __attribute__((ext_vector_type(4))) int v4i;

__device__ __forceinline__ unsigned lds_off(const void* p) {
  return (unsigned)(unsigned long long)(uintptr_t)p;   // low 32 bits of generic ptr = LDS byte offset
}

// 2D TDM tile load: tile_rows x 32 bf16 elements from a row-major (tensor_rows x Kdim)
// bf16 tensor into LDS, padding +4 DWORDs per 64B row (LDS row stride = 80B = 40 bf16).
__device__ __forceinline__ void tdm_tile_load(const __bf16* gp, unsigned ldsoff,
                                              int tile_rows, int Kdim, int tensor_rows) {
  unsigned long long ga = (unsigned long long)(uintptr_t)gp;
  v4u g0;
  g0[0] = 1u;                                            // count=1, user descriptor
  g0[1] = ldsoff;                                        // lds_addr
  g0[2] = (unsigned)ga;                                  // global_addr[31:0]
  g0[3] = (unsigned)((ga >> 32) & 0x1FFFFFFu) | (2u << 30); // global_addr[56:32] | type=2
  v8i g1;
  g1[0] = (int)((1u << 16)      // data_size = 1 (2 bytes)
              | (1u << 20)      // pad_enable
              | (3u << 22)      // pad_interval: every 16 DWORDs (64B)
              | (3u << 25));    // pad_amount: 4 DWORDs (16B)
  g1[1] = (int)((((unsigned)Kdim & 0xFFFFu) << 16));                         // tensor_dim0[15:0]
  g1[2] = (int)((((unsigned)Kdim >> 16) & 0xFFFFu)
              | (((unsigned)tensor_rows & 0xFFFFu) << 16));                  // dim0 hi | dim1 lo
  g1[3] = (int)(((((unsigned)tensor_rows) >> 16) & 0xFFFFu) | (32u << 16));  // dim1 hi | tile_dim0=32
  g1[4] = (int)((unsigned)tile_rows & 0xFFFFu);                              // tile_dim1 (tile_dim2=0)
  g1[5] = Kdim;                                                              // tensor_dim0_stride
  g1[6] = 0;
  g1[7] = 0;
  v4i z4 = {0, 0, 0, 0};
  v8i z8 = {0, 0, 0, 0, 0, 0, 0, 0};
  // 6-arg toolchain form: (g0, g1, g2, g3, g_extra, cpol)
  __builtin_amdgcn_tensor_load_to_lds(g0, g1, z4, z4, z8, 0);
}
#endif

// ============================================================================
// Batched WMMA GEMM:  C[b] = act( A[b] (MxK, bf16) * W[b]^T (N rows of K, bf16) + bias )
// Block: 256 threads (8 waves); tile 128(M) x 128(N); K step 32; double-buffered LDS
// fed by the Tensor Data Mover (wave 0 issues, TENSORcnt + barrier synchronize).
// Wave layout: 4 waves in M x 2 waves in N; each wave: 32x64 strip = 8 WMMA / K-step,
// all fragments (2 A + 4 B) loaded first so the 8 WMMAs issue back-to-back.
// ============================================================================
__global__ __launch_bounds__(256) void gemm_wmma_kernel(
    const __bf16* __restrict__ A, const __bf16* __restrict__ W,
    const float* __restrict__ bias, float* __restrict__ C,
    int M, int N, int K,
    long long sA, long long sW, long long sC, int act)
{
  A += (long long)blockIdx.z * sA;
  W += (long long)blockIdx.z * sW;
  C += (long long)blockIdx.z * sC;

  __shared__ __bf16 As[2][128][40];   // 128 x 32 K, row stride 40 (80B, 16B-aligned rows)
  __shared__ __bf16 Ws[2][128][40];   // 128 x 32 K

  const int tid  = threadIdx.x;
  const int lane = tid & 31;
  const int wave = tid >> 5;
  const int wm   = (wave & 3) << 5;   // 0,32,64,96 within 128 (M)
  const int wn   = (wave >> 2) << 6;  // 0 or 64 within 128 (N)
  const int m0   = blockIdx.y * 128;
  const int n0   = blockIdx.x * 128;

  v8f acc[2][4] = {};

#ifndef HAVE_TDM
  auto load_tiles = [&](int k0, int buf) {
    for (int i = tid; i < 128 * 32; i += 256) {
      int r = i >> 5, c = i & 31;
      As[buf][r][c] = A[(long long)(m0 + r) * K + k0 + c];
      Ws[buf][r][c] = W[(long long)(n0 + r) * K + k0 + c];
    }
  };
#endif

  const bool issuer = (tid < 32);
  (void)issuer;

#ifdef HAVE_TDM
  if (issuer) {
    tdm_tile_load(A + (long long)m0 * K, lds_off(&As[0][0][0]), 128, K, M);
    tdm_tile_load(W + (long long)n0 * K, lds_off(&Ws[0][0][0]), 128, K, N);
    __builtin_amdgcn_s_wait_tensorcnt((short)0);
  }
  __syncthreads();
#else
  load_tiles(0, 0);
  __syncthreads();
#endif

  const int ln15 = lane & 15;
  const int kb   = (lane < 16) ? 0 : 8;

  int nb = 0;
  for (int k0 = 0; k0 < K; k0 += 32) {
    const int nxt = nb ^ 1;
#ifdef HAVE_TDM
    if (k0 + 32 < K && issuer) {
      tdm_tile_load(A + (long long)m0 * K + (k0 + 32), lds_off(&As[nxt][0][0]), 128, K, M);
      tdm_tile_load(W + (long long)n0 * K + (k0 + 32), lds_off(&Ws[nxt][0][0]), 128, K, N);
    }
#else
    if (k0 + 32 < K) load_tiles(k0 + 32, nxt);
#endif

    // Fragment layout: CDNA5 16-bit A 16x32
    // lanes 0-15 : M=lane,    VGPR0..3 -> K 0..7,  VGPR4..7 -> K 16..23
    // lanes16-31 : M=lane-16, VGPR0..3 -> K 8..15, VGPR4..7 -> K 24..31
    v16bf af[2], bf[4];
    #pragma unroll
    for (int h = 0; h < 2; ++h) {
      const int mr = wm + h * 16 + ln15;
      #pragma unroll
      for (int i = 0; i < 8; ++i) {
        int kk = kb + ((i < 4) ? (2 * i) : (16 + 2 * (i - 4)));
        af[h][2 * i]     = As[nb][mr][kk];
        af[h][2 * i + 1] = As[nb][mr][kk + 1];
      }
    }
    #pragma unroll
    for (int t = 0; t < 4; ++t) {
      const int nr = wn + t * 16 + ln15;
      #pragma unroll
      for (int i = 0; i < 8; ++i) {
        int kk = kb + ((i < 4) ? (2 * i) : (16 + 2 * (i - 4)));
        bf[t][2 * i]     = Ws[nb][nr][kk];
        bf[t][2 * i + 1] = Ws[nb][nr][kk + 1];
      }
    }
    // 8 back-to-back WMMAs; every A fragment used 4x, every B fragment used 2x
    #pragma unroll
    for (int t = 0; t < 4; ++t) {
      acc[0][t] = __builtin_amdgcn_wmma_f32_16x16x32_bf16(false, af[0], false, bf[t],
                                                          (short)0, acc[0][t], false, false);
      acc[1][t] = __builtin_amdgcn_wmma_f32_16x16x32_bf16(false, af[1], false, bf[t],
                                                          (short)0, acc[1][t], false, false);
    }

#ifdef HAVE_TDM
    if (issuer) __builtin_amdgcn_s_wait_tensorcnt((short)0);
#endif
    __syncthreads();
    nb = nxt;
  }

  // C/D layout: lanes 0-15 -> N=lane, rows M 0..7; lanes 16-31 -> N=lane-16, rows M 8..15
  #pragma unroll
  for (int h = 0; h < 2; ++h) {
    const int mb = m0 + wm + h * 16 + ((lane < 16) ? 0 : 8);
    #pragma unroll
    for (int t = 0; t < 4; ++t) {
      const int nn = n0 + wn + t * 16 + ln15;
      const float bv = bias[nn];          // always valid (zero vector when no bias)
      #pragma unroll
      for (int j = 0; j < 8; ++j) {
        float v = acc[h][t][j] + bv;
        if (act == 1) v = 0.5f * v * (1.0f + erff(v * 0.70710678118654752f)); // exact GELU
        C[(long long)(mb + j) * N + nn] = v;
      }
    }
  }
}

// one-time fp32 -> bf16 conversion (bandwidth-bound)
__global__ void cvt_bf16_kernel(const float* __restrict__ in, __bf16* __restrict__ outp,
                                long long n) {
  long long i = (long long)blockIdx.x * 256 + threadIdx.x;
  if (i < n) outp[i] = (__bf16)in[i];
}

// ============================================================================
// Small support kernels (VALU; bandwidth-trivial parts of the model)
// ============================================================================
__global__ void pos_embed_kernel(float* __restrict__ pe) {
  int i = blockIdx.x * 256 + threadIdx.x;
  if (i >= LL * DM) return;
  int d = i % DM, l = i / DM;
  int i2 = d >> 1;
  float div = expf((float)(2 * i2) * (-logf(10000.0f) / (float)DM));
  float a = (float)l * div;
  pe[i] = (d & 1) ? cosf(a) : sinf(a);
}

// circular 3-tap conv embedding: out[b,l,o] = sum_{kk,c} x[b,(l-1+kk)%L,c]*W[o,c,kk] + pe[l,o]
__global__ void embed_kernel(const float* __restrict__ x, const float* __restrict__ Wt,
                             const float* __restrict__ pe, float* __restrict__ out) {
  int i = blockIdx.x * 256 + threadIdx.x;
  if (i >= BB * LL * DM) return;
  int o = i % DM; int l = (i / DM) % LL; int b = i / (DM * LL);
  float s = 0.0f;
  #pragma unroll
  for (int kk = 0; kk < 3; ++kk) {
    int ls = (l - 1 + kk + LL) % LL;
    const float* xr = x + ((long long)b * LL + ls) * CIN;
    const float* wr = Wt + o * (CIN * 3) + kk;
    #pragma unroll
    for (int c = 0; c < CIN; ++c) s += xr[c] * wr[c * 3];
  }
  out[i] = s + pe[l * DM + o];
}

// series decomposition of (x [+ y]) with 25-tap replicate-padded moving mean.
__global__ void decomp_kernel(const float* __restrict__ x, const float* __restrict__ y,
                              float* __restrict__ season, float* __restrict__ trend,
                              int accum, int Cc, int total) {
  int i = blockIdx.x * 256 + threadIdx.x;
  if (i >= total) return;
  int c = i % Cc; int l = (i / Cc) % LL; long long b = i / (Cc * LL);
  const long long rowb = b * LL;
  float s = 0.0f;
  #pragma unroll 5
  for (int j = -12; j <= 12; ++j) {
    int ll = l + j; ll = ll < 0 ? 0 : (ll > LL - 1 ? LL - 1 : ll);
    long long id = (rowb + ll) * Cc + c;
    float v = x[id]; if (y) v += y[id];
    s += v;
  }
  float m = s * (1.0f / 25.0f);
  long long id0 = (rowb + l) * Cc + c;
  float ctr = x[id0] + (y ? y[id0] : 0.0f);
  if (season) season[id0] = ctr - m;
  if (trend) { if (accum) trend[id0] += m; else trend[id0] = m; }
}

__global__ void mean_time_kernel(const float* __restrict__ x, float* __restrict__ m) {
  int i = blockIdx.x * 256 + threadIdx.x;
  if (i >= BB * CIN) return;
  int c = i % CIN; int b = i / CIN;
  float s = 0.0f;
  for (int t = 0; t < LL; ++t) s += x[((long long)b * LL + t) * CIN + c];
  m[i] = s / (float)LL;
}

__global__ void build_inits_kernel(const float* __restrict__ sea, const float* __restrict__ trd,
                                   const float* __restrict__ xm,
                                   float* __restrict__ sinit, float* __restrict__ trend) {
  int i = blockIdx.x * 256 + threadIdx.x;
  if (i >= BB * LL * CIN) return;
  int c = i % CIN; int t = (i / CIN) % LL; int b = i / (CIN * LL);
  if (t < LABL) {
    long long id = ((long long)b * LL + (LL - LABL + t)) * CIN + c;
    sinit[i] = sea[id];
    trend[i] = trd[id];
  } else {
    sinit[i] = 0.0f;
    trend[i] = xm[b * CIN + c];
  }
}

// mean_value[b,tau] = (1/512) * sum_t M[b, t, (t - tau) mod L]
__global__ void diag_corr_kernel(const float* __restrict__ Mc, float* __restrict__ mv) {
  int b = blockIdx.x, tau = threadIdx.x;
  const float* Mb = Mc + (long long)b * LL * LL;
  float s = 0.0f;
  for (int t = 0; t < LL; ++t) {
    int ss = t - tau; if (ss < 0) ss += LL;
    s += Mb[t * LL + ss];
  }
  mv[b * LL + tau] = s * (1.0f / (float)DM);
}

__global__ void batch_mean_kernel(const float* __restrict__ mv, float* __restrict__ g) {
  int tau = threadIdx.x;
  float s = 0.0f;
  for (int b = 0; b < BB; ++b) s += mv[b * LL + tau];
  g[tau] = s / (float)BB;
}

__global__ void topk_kernel(const float* __restrict__ g, int* __restrict__ idx) {
  if (threadIdx.x != 0 || blockIdx.x != 0) return;
  float loc[LL];
  for (int i = 0; i < LL; ++i) loc[i] = g[i];
  for (int j = 0; j < TKK; ++j) {
    float best = -1e30f; int bi = 0;
    for (int i = 0; i < LL; ++i) if (loc[i] > best) { best = loc[i]; bi = i; }
    idx[j] = bi; loc[bi] = -1e30f;
  }
}

__global__ void softmax_w_kernel(const float* __restrict__ mv, const int* __restrict__ idx,
                                 float* __restrict__ w) {
  if (threadIdx.x != 0) return;
  int b = blockIdx.x;
  float v[TKK]; float mx = -1e30f;
  for (int j = 0; j < TKK; ++j) { v[j] = mv[b * LL + idx[j]]; mx = fmaxf(mx, v[j]); }
  float s = 0.0f;
  for (int j = 0; j < TKK; ++j) { v[j] = expf(v[j] - mx); s += v[j]; }
  for (int j = 0; j < TKK; ++j) w[b * TKK + j] = v[j] / s;
}

// out[b,t,d] = sum_j w[b,j] * v[b, (t + idx_j) mod L, d]
__global__ void agg_kernel(const float* __restrict__ v, const float* __restrict__ w,
                           const int* __restrict__ idx, float* __restrict__ out) {
  int i = blockIdx.x * 256 + threadIdx.x;
  if (i >= BB * LL * DM) return;
  int d = i % DM; int t = (i / DM) % LL; int b = i / (DM * LL);
  float s = 0.0f;
  #pragma unroll
  for (int j = 0; j < TKK; ++j) {
    int tt = t + idx[j]; if (tt >= LL) tt -= LL;
    s += w[b * TKK + j] * v[((long long)b * LL + tt) * DM + d];
  }
  out[i] = s;
}

// trend[b,l,o] += sum_{kk,c} tsum[b,(l-1+kk)%L,c] * Wt[o,c,kk]
__global__ void trend_conv_kernel(const float* __restrict__ ts, const float* __restrict__ Wt,
                                  float* __restrict__ trend) {
  int i = blockIdx.x * 256 + threadIdx.x;
  if (i >= BB * LL * CIN) return;
  int o = i % CIN; int l = (i / CIN) % LL; int b = i / (CIN * LL);
  float s = 0.0f;
  #pragma unroll
  for (int kk = 0; kk < 3; ++kk) {
    int ls = (l - 1 + kk + LL) % LL;
    const float* xr = ts + ((long long)b * LL + ls) * DM;
    const float* wr = Wt + o * (DM * 3) + kk;
    for (int c = 0; c < DM; ++c) s += xr[c] * wr[c * 3];
  }
  trend[i] += s;
}

// rowwise LayerNorm over D=512 (biased var), scale/shift
__global__ void ln_rows_kernel(const float* __restrict__ x, const float* __restrict__ g,
                               const float* __restrict__ bta, float* __restrict__ out) {
  __shared__ float red[256];
  int row = blockIdx.x; int tid = threadIdx.x;
  const float* xr = x + (long long)row * DM;
  float a0 = xr[tid], a1 = xr[tid + 256];
  red[tid] = a0 + a1; __syncthreads();
  for (int s = 128; s > 0; s >>= 1) { if (tid < s) red[tid] += red[tid + s]; __syncthreads(); }
  float mu = red[0] / (float)DM; __syncthreads();
  float d0 = a0 - mu, d1 = a1 - mu;
  red[tid] = d0 * d0 + d1 * d1; __syncthreads();
  for (int s = 128; s > 0; s >>= 1) { if (tid < s) red[tid] += red[tid + s]; __syncthreads(); }
  float inv = rsqrtf(red[0] / (float)DM + 1e-5f);
  out[(long long)row * DM + tid]       = d0 * inv * g[tid]       + bta[tid];
  out[(long long)row * DM + tid + 256] = d1 * inv * g[tid + 256] + bta[tid + 256];
}

// subtract per-(b,d) mean over time axis
__global__ void ln_subtime_kernel(float* __restrict__ x) {
  int i = blockIdx.x * 256 + threadIdx.x;
  if (i >= BB * DM) return;
  int d = i % DM; int b = i / DM;
  float s = 0.0f;
  for (int t = 0; t < LL; ++t) s += x[((long long)b * LL + t) * DM + d];
  float mu = s / (float)LL;
  for (int t = 0; t < LL; ++t) x[((long long)b * LL + t) * DM + d] -= mu;
}

// out[b,t,o] = trend[b,192+t,o] + proj_b[o] + xh[b,192+t,:] . proj_W[o,:]
__global__ void final_kernel(const float* __restrict__ xh, const float* __restrict__ trend,
                             const float* __restrict__ pW, const float* __restrict__ pb,
                             float* __restrict__ out) {
  int i = blockIdx.x * 256 + threadIdx.x;
  if (i >= BB * PRD * CIN) return;
  int o = i % CIN; int t = (i / CIN) % PRD; int b = i / (CIN * PRD);
  int ft = LL - PRD + t;
  float s = trend[((long long)b * LL + ft) * CIN + o] + pb[o];
  const float* xr = xh + ((long long)b * LL + ft) * DM;
  const float* wr = pW + o * DM;
  for (int d = 0; d < DM; ++d) s += xr[d] * wr[d];
  out[i] = s;
}

__global__ void zero_kernel(float* __restrict__ p, long long n) {
  long long i = (long long)blockIdx.x * 256 + threadIdx.x;
  if (i < n) p[i] = 0.0f;
}

// ============================================================================
// Orchestration
// ============================================================================
extern "C" void kernel_launch(void* const* d_in, const int* in_sizes, int n_in,
                              void* d_out, int out_size, void* d_ws, size_t ws_size,
                              hipStream_t stream)
{
  (void)in_sizes; (void)n_in; (void)out_size; (void)ws_size;

  const float* x_enc     = (const float*)d_in[0];
  const float* enc_emb_W = (const float*)d_in[2];
  const float* dec_emb_W = (const float*)d_in[3];
  const float* enc_Wq = (const float*)d_in[4];  const float* enc_bq = (const float*)d_in[5];
  const float* enc_Wk = (const float*)d_in[6];  const float* enc_bk = (const float*)d_in[7];
  const float* enc_Wv = (const float*)d_in[8];  const float* enc_bv = (const float*)d_in[9];
  const float* enc_Wo = (const float*)d_in[10]; const float* enc_bo = (const float*)d_in[11];
  const float* enc_W1 = (const float*)d_in[12]; const float* enc_W2 = (const float*)d_in[13];
  const float* dsWq = (const float*)d_in[14];  const float* dsbq = (const float*)d_in[15];
  const float* dsWk = (const float*)d_in[16];  const float* dsbk = (const float*)d_in[17];
  const float* dsWv = (const float*)d_in[18];  const float* dsbv = (const float*)d_in[19];
  const float* dsWo = (const float*)d_in[20];  const float* dsbo = (const float*)d_in[21];
  const float* dcWq = (const float*)d_in[22];  const float* dcbq = (const float*)d_in[23];
  const float* dcWk = (const float*)d_in[24];  const float* dcbk = (const float*)d_in[25];
  const float* dcWv = (const float*)d_in[26];  const float* dcbv = (const float*)d_in[27];
  const float* dcWo = (const float*)d_in[28];  const float* dcbo = (const float*)d_in[29];
  const float* dec_W1 = (const float*)d_in[30]; const float* dec_W2 = (const float*)d_in[31];
  const float* dec_Wt = (const float*)d_in[32];
  const float* norm_g = (const float*)d_in[33]; const float* norm_b = (const float*)d_in[34];
  const float* proj_W = (const float*)d_in[35]; const float* proj_b = (const float*)d_in[36];
  float* out = (float*)d_out;

  const long long ML = (long long)BB * LL * DM;     // 3,145,728
  float* ws = (float*)d_ws;
  size_t off = 0;
  auto alloc = [&](long long n) { float* p = ws + off; off += (size_t)n; return p; };

  float* pe    = alloc((long long)LL * DM);
  float* encx  = alloc(ML);
  float* decx  = alloc(ML);
  float* qb    = alloc(ML);
  float* kb    = alloc(ML);
  float* vb    = alloc(ML);
  float* t1    = alloc(ML);
  float* t2    = alloc(ML);
  float* tsum  = alloc(ML);
  float* ffnh  = alloc((long long)BB * LL * DFF);
  float* corrM = alloc((long long)BB * LL * LL);
  float* mv    = alloc(BB * LL);
  float* gbuf  = alloc(LL);
  float* wbuf  = alloc(BB * TKK);
  float* seaR  = alloc((long long)BB * LL * CIN);
  float* trdR  = alloc((long long)BB * LL * CIN);
  float* sinit = alloc((long long)BB * LL * CIN);
  float* trend = alloc((long long)BB * LL * CIN);
  float* xm    = alloc(BB * CIN);
  float* zbias = alloc(DFF);                                         // always-valid zero bias
  __bf16* Abf  = (__bf16*)alloc((long long)BB * LL * DFF / 2 + 16);  // max-A bf16 staging
  __bf16* Wbf  = (__bf16*)alloc(ML / 2 + 16);                        // max-W bf16 staging
  int*   idx   = (int*)(ws + off); off += 8;

  const int Mfull = BB * LL;  // 6144
  auto grid1 = [](long long n) { return dim3((unsigned)((n + 255) / 256)); };
  auto gemm = [&](const float* A, const float* Wm, const float* bias, float* Cm,
                  int M, int N, int K, int batch,
                  long long sA, long long sW, long long sC, int act) {
    long long nA = (long long)(batch - 1) * sA + (long long)M * K;
    long long nW = (long long)(batch - 1) * sW + (long long)N * K;
    cvt_bf16_kernel<<<grid1(nA), 256, 0, stream>>>(A, Abf, nA);
    cvt_bf16_kernel<<<grid1(nW), 256, 0, stream>>>(Wm, Wbf, nW);
    dim3 grid(N / 128, M / 128, batch);
    gemm_wmma_kernel<<<grid, 256, 0, stream>>>(Abf, Wbf, bias ? bias : zbias, Cm,
                                               M, N, K, sA, sW, sC, act);
  };

  // --- preamble ---
  zero_kernel<<<grid1(DFF), 256, 0, stream>>>(zbias, DFF);
  pos_embed_kernel<<<grid1((long long)LL * DM), 256, 0, stream>>>(pe);
  mean_time_kernel<<<grid1(BB * CIN), 256, 0, stream>>>(x_enc, xm);
  decomp_kernel<<<grid1((long long)BB * LL * CIN), 256, 0, stream>>>(
      x_enc, nullptr, seaR, trdR, 0, CIN, BB * LL * CIN);
  build_inits_kernel<<<grid1((long long)BB * LL * CIN), 256, 0, stream>>>(
      seaR, trdR, xm, sinit, trend);
  embed_kernel<<<grid1(ML), 256, 0, stream>>>(x_enc, enc_emb_W, pe, encx);

  // --- AutoCorrelation attention block (qkv -> corr -> topk -> agg -> Wo) ---
  auto attn = [&](const float* xq, const float* xkv,
                  const float* Wq, const float* bq_, const float* Wk_, const float* bk_,
                  const float* Wv_, const float* bv_, const float* Wo_, const float* bo_,
                  float* aggbuf, float* outbuf) {
    gemm(xq,  Wq,  bq_, qb, Mfull, DM, DM, 1, 0, 0, 0, 0);
    gemm(xkv, Wk_, bk_, kb, Mfull, DM, DM, 1, 0, 0, 0, 0);
    gemm(xkv, Wv_, bv_, vb, Mfull, DM, DM, 1, 0, 0, 0, 0);
    // batched Q K^T  (replaces the FFT correlation; head-mean folds into /512)
    gemm(qb, kb, nullptr, corrM, LL, LL, DM, BB,
         (long long)LL * DM, (long long)LL * DM, (long long)LL * LL, 0);
    diag_corr_kernel<<<BB, LL, 0, stream>>>(corrM, mv);
    batch_mean_kernel<<<1, LL, 0, stream>>>(mv, gbuf);
    topk_kernel<<<1, 1, 0, stream>>>(gbuf, idx);
    softmax_w_kernel<<<BB, 1, 0, stream>>>(mv, idx, wbuf);
    agg_kernel<<<grid1(ML), 256, 0, stream>>>(vb, wbuf, idx, aggbuf);
    gemm(aggbuf, Wo_, bo_, outbuf, Mfull, DM, DM, 1, 0, 0, 0, 0);
  };

  // --- encoder: 2 layers ---
  for (int l = 0; l < 2; ++l) {
    attn(encx, encx,
         enc_Wq + (long long)l * DM * DM, enc_bq + l * DM,
         enc_Wk + (long long)l * DM * DM, enc_bk + l * DM,
         enc_Wv + (long long)l * DM * DM, enc_bv + l * DM,
         enc_Wo + (long long)l * DM * DM, enc_bo + l * DM, t1, t2);
    decomp_kernel<<<grid1(ML), 256, 0, stream>>>(encx, t2, t1, nullptr, 0, DM, (int)ML);
    gemm(t1, enc_W1 + (long long)l * DFF * DM, nullptr, ffnh, Mfull, DFF, DM, 1, 0, 0, 0, 1);
    gemm(ffnh, enc_W2 + (long long)l * DM * DFF, nullptr, t2, Mfull, DM, DFF, 1, 0, 0, 0, 0);
    decomp_kernel<<<grid1(ML), 256, 0, stream>>>(t1, t2, encx, nullptr, 0, DM, (int)ML);
  }

  // --- decoder: 1 layer ---
  embed_kernel<<<grid1(ML), 256, 0, stream>>>(sinit, dec_emb_W, pe, decx);
  zero_kernel<<<grid1(ML), 256, 0, stream>>>(tsum, ML);

  // self-attention (x = decx)
  attn(decx, decx, dsWq, dsbq, dsWk, dsbk, dsWv, dsbv, dsWo, dsbo, t1, t2);
  decomp_kernel<<<grid1(ML), 256, 0, stream>>>(decx, t2, t1, tsum, 1, DM, (int)ML);
  // cross-attention (x = t1, kv = encoder output)
  attn(t1, encx, dcWq, dcbq, dcWk, dcbk, dcWv, dcbv, dcWo, dcbo, decx, t2);
  decomp_kernel<<<grid1(ML), 256, 0, stream>>>(t1, t2, decx, tsum, 1, DM, (int)ML);
  // FFN (x = decx)
  gemm(decx, dec_W1, nullptr, ffnh, Mfull, DFF, DM, 1, 0, 0, 0, 1);
  gemm(ffnh, dec_W2, nullptr, t2, Mfull, DM, DFF, 1, 0, 0, 0, 0);
  decomp_kernel<<<grid1(ML), 256, 0, stream>>>(decx, t2, t1, tsum, 1, DM, (int)ML);
  // trend projection of (t1+t2+t3)
  trend_conv_kernel<<<grid1((long long)BB * LL * CIN), 256, 0, stream>>>(tsum, dec_Wt, trend);

  // --- final norm + projection + slice ---
  ln_rows_kernel<<<BB * LL, 256, 0, stream>>>(t1, norm_g, norm_b, t2);
  ln_subtime_kernel<<<grid1(BB * DM), 256, 0, stream>>>(t2);
  final_kernel<<<grid1((long long)BB * PRD * CIN), 256, 0, stream>>>(t2, trend, proj_W, proj_b, out);
}
